// DTWLoss_52450140619396
// MI455X (gfx1250) — matrix operand order repaired and checked by reading, not compiled, not run on hardware
//
#include <hip/hip_runtime.h>
#include <hip/hip_bf16.h>

// ---------------------------------------------------------------------------
// Soft-DTW triplet loss for MI455X (gfx1250, wave32, WMMA).
//
// Pipeline (all on `stream`):
//   K1 normalize_rows : fp32 [8,1024,64] -> unit-norm f16 rows in ws
//   K2 cost_gemm      : 5 pairs x 8 batches, C = 1 - An.Bn^T via
//                       v_wmma_f32_16x16x32_f16. Each wave computes a 16x64
//                       strip (A fragment reused over 4 B tiles -> 8 wmma).
//                       Cost stored f16 in anti-diagonal-major layout
//                       cost[k][i], k = i + j, k padded to 2048.
//   K3 softdtw_dp     : 40 blocks x 1024 threads, 2047 barrier-stepped
//                       anti-diagonals, LDS triple-buffered, coalesced f16
//                       cost loads + next-diagonal global_prefetch.
//   K4 finalize       : MSE over batch -> out[0]
//
// Workspace layout (halves unless noted):
//   [0 , R)     TGTn   R = 8*1024*64 = 524288
//   [R , 2R)    OTHn
//   [2R, 3R)    Xn
//   [3R, 3R + 40*2048*1024)  cost diagonals (f16)   ~167.8 MB
//   then 40 floats S                                 total ~171 MB
// ---------------------------------------------------------------------------

typedef __attribute__((ext_vector_type(16))) _Float16 v16h;
typedef __attribute__((ext_vector_type(8)))  _Float16 v8h;
typedef __attribute__((ext_vector_type(8)))  float    v8f;

#define RELEMS (8 * 1024 * 64)       // halves per tensor
#define KPAD   2048                  // padded diagonal count
#define NSEQ   1024
#define INFV   1e10f

// ------------------------- K1: normalize rows ------------------------------
__global__ void dtw_normalize(const float* __restrict__ TGT,
                              const float* __restrict__ OTH,
                              const float* __restrict__ X,
                              _Float16* __restrict__ nb) {
  const int wave = threadIdx.x >> 5;
  const int lane = threadIdx.x & 31;
  const int row  = blockIdx.x * 8 + wave;        // 0 .. 3*8192-1
  const int t    = row >> 13;                    // tensor id 0..2
  const int rr   = row & 8191;                   // row in tensor

  const float* src = (t == 0) ? TGT : (t == 1) ? OTH : X;
  src += (size_t)rr * 64;
  _Float16* dst = nb + (size_t)t * RELEMS + (size_t)rr * 64;

  float x0 = src[lane];
  float x1 = src[lane + 32];
  float ss = x0 * x0 + x1 * x1;
#pragma unroll
  for (int o = 16; o; o >>= 1) ss += __shfl_xor(ss, o, 32);
  float scale = 1.0f / fmaxf(sqrtf(ss), 1e-8f);
  dst[lane]      = (_Float16)(x0 * scale);
  dst[lane + 32] = (_Float16)(x1 * scale);
}

// ------------------------- K2: WMMA cost GEMM ------------------------------
// grid = (128, 8, 5), block = 256 (8 waves).
// Each wave: one 16x64 strip of C (4 tiles of 16x16), A fragment reused.
__global__ void dtw_cost_gemm(const _Float16* __restrict__ nb,
                              _Float16* __restrict__ cost) {
  const int p = blockIdx.z;              // pair id
  const int b = blockIdx.y;              // batch
  const int wave = threadIdx.x >> 5;
  const int lane = threadIdx.x & 31;
  const int job  = blockIdx.x * 8 + wave;          // 0..1023
  const int ti   = job >> 4;                       // 0..63   (row tile)
  const int j00  = (job & 15) << 6;                // 0..960  (strip base col)
  const int i0   = ti << 4;

  // pair -> (A tensor, B tensor) in units of RELEMS: X=2, TGT=0, OTH=1
  const int aoff[5] = {2, 0, 0, 1, 1};
  const int boff[5] = {2, 2, 0, 2, 1};
  const _Float16* A = nb + (size_t)aoff[p] * RELEMS + (size_t)b * NSEQ * 64;
  const _Float16* B = nb + (size_t)boff[p] * RELEMS + (size_t)b * NSEQ * 64;

  const int hi = lane >> 4;              // 0 for lanes 0-15, 1 for 16-31
  const int lr = lane & 15;

  // A fragments (16x32 f16) for kb = 0 and kb = 32:
  // lane<16 holds K kb..kb+7 (V0-3) and kb+16..kb+23 (V4-7);
  // lanes 16-31 the +8 complement.
  const _Float16* arow = A + (size_t)(i0 + lr) * 64;
  const int ka0 = (hi << 3);
  v8h a0lo = *(const v8h*)(arow + ka0);
  v8h a0hi = *(const v8h*)(arow + ka0 + 16);
  v16h af0 = __builtin_shufflevector(a0lo, a0hi, 0, 1, 2, 3, 4, 5, 6, 7,
                                     8, 9, 10, 11, 12, 13, 14, 15);
  v8h a1lo = *(const v8h*)(arow + 32 + ka0);
  v8h a1hi = *(const v8h*)(arow + 32 + ka0 + 16);
  v16h af1 = __builtin_shufflevector(a1lo, a1hi, 0, 1, 2, 3, 4, 5, 6, 7,
                                     8, 9, 10, 11, 12, 13, 14, 15);

  // Per-lane diagonal store bases: k*1024 + i = i*1025 + j*1024.
  // r-th output row i_r = i0 + r + 8*hi, col j = j00 + lr (+16 per tile).
  _Float16* costp = cost + (size_t)(p * 8 + b) * KPAD * NSEQ;
  const int n = lr;
  size_t base[8];
#pragma unroll
  for (int r = 0; r < 8; ++r) {
    int i = i0 + r + (hi << 3);
    base[r] = (size_t)i * (NSEQ + 1) + (size_t)(j00 + n) * NSEQ;
  }

#pragma unroll
  for (int t = 0; t < 4; ++t) {
    const int j0 = j00 + (t << 4);
    // B fragment (32x16 f16, NT): lane<16 holds K kb..kb+7 and kb+16..kb+23
    // of row j0+lr, lanes 16-31 the +8 complement (same order as A).
    const _Float16* brow = B + (size_t)(j0 + lr) * 64;
    const int kb0 = (hi << 3);
    v8h b0lo = *(const v8h*)(brow + kb0);
    v8h b0hi = *(const v8h*)(brow + kb0 + 16);
    v16h bf0 = __builtin_shufflevector(b0lo, b0hi, 0, 1, 2, 3, 4, 5, 6, 7,
                                       8, 9, 10, 11, 12, 13, 14, 15);
    v8h b1lo = *(const v8h*)(brow + 32 + kb0);
    v8h b1hi = *(const v8h*)(brow + 32 + kb0 + 16);
    v16h bf1 = __builtin_shufflevector(b1lo, b1hi, 0, 1, 2, 3, 4, 5, 6, 7,
                                       8, 9, 10, 11, 12, 13, 14, 15);

    v8f c = {};
    c = __builtin_amdgcn_wmma_f32_16x16x32_f16(
        false, af0, false, bf0, (short)0, c, false, false);
    c = __builtin_amdgcn_wmma_f32_16x16x32_f16(
        false, af1, false, bf1, (short)0, c, false, false);

    const size_t toff = (size_t)(t << 4) * NSEQ;   // +16 cols = +16*1024 halves
#pragma unroll
    for (int r = 0; r < 8; ++r) {
      costp[base[r] + toff] = (_Float16)(1.0f - c[r]);
    }
  }
}

// ------------------------- K3: soft-DTW DP ---------------------------------
// grid = 40, block = 1024. gamma = 1.
__global__ void dtw_dp(const _Float16* __restrict__ cost,
                       float* __restrict__ S) {
  __shared__ float buf[3][NSEQ];
  const int pb = blockIdx.x;
  const _Float16* costp = cost + (size_t)pb * KPAD * NSEQ;
  const int i = threadIdx.x;

  buf[0][i] = INFV;
  buf[1][i] = INFV;
  buf[2][i] = INFV;
  __syncthreads();

  for (int k = 0; k < 2 * NSEQ - 1; ++k) {
    float* cur        = buf[k % 3];
    const float* d1   = buf[(k + 2) % 3];   // diag k-1
    const float* d2   = buf[(k + 1) % 3];   // diag k-2
    // Next diagonal line is always inside the padded slab; get it in flight
    // while this step's exp/log chain and barrier run.
    __builtin_prefetch(costp + (size_t)(k + 1) * NSEQ + i, 0, 1);
    const int j = k - i;
    float v = INFV;
    if (j >= 0 && j < NSEQ) {
      float r2  = (i == 0) ? ((k == 0) ? 0.0f : INFV) : d2[i - 1];
      float r1a = (i == 0) ? INFV : d1[i - 1];
      float r1b = d1[i];
      float m = fminf(r2, fminf(r1a, r1b));
      float s = __expf(m - r2) + __expf(m - r1a) + __expf(m - r1b);
      float softmin = m - __logf(s);
      float cc = (float)costp[(size_t)k * NSEQ + i];
      v = cc + softmin;
      if (k == 2 * NSEQ - 2 && i == NSEQ - 1) S[pb] = v;  // R[N-1, M-1]
    }
    cur[i] = v;
    __syncthreads();
  }
}

// ------------------------- K4: finalize ------------------------------------
__global__ void dtw_finalize(const float* __restrict__ S,
                             const float* __restrict__ labels,
                             float* __restrict__ out) {
  const int b = threadIdx.x;
  float v = 0.0f;
  if (b < 8) {
    float sxx  = S[0 * 8 + b];
    float dtgt = S[1 * 8 + b] - 0.5f * (S[2 * 8 + b] + sxx);
    float doth = S[3 * 8 + b] - 0.5f * (S[4 * 8 + b] + sxx);
    float t = dtgt - doth - labels[0];
    v = t * t;
  }
#pragma unroll
  for (int o = 16; o; o >>= 1) v += __shfl_down(v, o, 32);
  if (b == 0) out[0] = v * (1.0f / 8.0f);
}

// ---------------------------------------------------------------------------
extern "C" void kernel_launch(void* const* d_in, const int* in_sizes, int n_in,
                              void* d_out, int out_size, void* d_ws, size_t ws_size,
                              hipStream_t stream) {
  const float* TGT    = (const float*)d_in[0];
  const float* OTH    = (const float*)d_in[1];
  const float* X      = (const float*)d_in[2];
  const float* labels = (const float*)d_in[3];
  float* out = (float*)d_out;

  _Float16* nb   = (_Float16*)d_ws;
  _Float16* cost = nb + (size_t)3 * RELEMS;
  float*    S    = (float*)(cost + (size_t)40 * KPAD * NSEQ);

  dtw_normalize<<<3072, 256, 0, stream>>>(TGT, OTH, X, nb);
  dtw_cost_gemm<<<dim3(128, 8, 5), 256, 0, stream>>>(nb, cost);
  dtw_dp<<<40, 1024, 0, stream>>>(cost, S);
  dtw_finalize<<<1, 32, 0, stream>>>(S, labels, out);
}
